// MultiHeadedAttention_89197880803679
// MI455X (gfx1250) — matrix operand order repaired and checked by reading, not compiled, not run on hardware
//
#include <hip/hip_runtime.h>
#include <hip/hip_bf16.h>
#include <math.h>

// Problem constants (from reference)
#define BATCH 4096
#define LVL   32
#define DM    512
#define NHEAD 8
#define DH    64
#define NB    2            // batch rows per fused attention block
#define APAD  8            // bf16 row padding (16B) for LDS bank spread
#define ASTR  (DM + APAD)  // 520 bf16 elements per row

// 256 threads = 8 waves; attention kernel uses ~206KB LDS -> 1 block/WGP -> 2 waves/SIMD.
// Tell the compiler so it allocates a big enough VGPR budget (no accumulator spills).
#define LB __launch_bounds__(256, 2)

typedef __attribute__((ext_vector_type(16))) __bf16 v16bf;
typedef __attribute__((ext_vector_type(8)))  float  v8f;
typedef __attribute__((ext_vector_type(4)))  unsigned int u32x4;
typedef __attribute__((ext_vector_type(4)))  float  f32x4;

union FragU { v16bf v; u32x4 q[2]; };

// A-fragment (16x32 bf16): lane m=lane&15, half=lane>>4; two 16B chunks at K = 8*half and 16+8*half
__device__ __forceinline__ v16bf load_fragA(const __bf16* base, int stride) {
  const int lane = threadIdx.x & 31;
  const __bf16* p = base + (lane & 15) * stride + (lane >> 4) * 8;
  FragU u;
  u.q[0] = *(const u32x4*)(p);
  u.q[1] = *(const u32x4*)(p + 16);
  return u.v;
}

// B-fragment (32x16 bf16) loaded from W^T rows: lane n=lane&15, half=lane>>4; two 16B chunks at K = 8*half and 16+8*half
__device__ __forceinline__ v16bf load_fragB(const __bf16* base, int stride) {
  const int lane = threadIdx.x & 31;
  const __bf16* p = base + (lane & 15) * stride + (lane >> 4) * 8;
  FragU u;
  u.q[0] = *(const u32x4*)(p);
  u.q[1] = *(const u32x4*)(p + 16);
  return u.v;
}

__device__ __forceinline__ v8f wmma_bf16(v16bf a, v16bf b, v8f c) {
  return __builtin_amdgcn_wmma_f32_16x16x32_bf16(false, a, false, b, (short)0, c, false, false);
}

__device__ __forceinline__ v8f zero8() { v8f z = {0.f,0.f,0.f,0.f,0.f,0.f,0.f,0.f}; return z; }

// ---------------- Kernel 0: W -> W^T in bf16 (4 matrices) ----------------
__global__ void wconvert_kernel(const float* __restrict__ Wq, const float* __restrict__ Wk,
                                const float* __restrict__ Wv, const float* __restrict__ Wo,
                                __bf16* __restrict__ WqT, __bf16* __restrict__ WkT,
                                __bf16* __restrict__ WvT, __bf16* __restrict__ WoT) {
  int idx = blockIdx.x * 256 + threadIdx.x;   // idx = n*512 + k  -> T[n][k] = W[k][n]
  int n = idx >> 9, k = idx & (DM - 1);
  int src = k * DM + n;
  WqT[idx] = (__bf16)Wq[src];
  WkT[idx] = (__bf16)Wk[src];
  WvT[idx] = (__bf16)Wv[src];
  WoT[idx] = (__bf16)Wo[src];
}

// ---------------- Kernel: fill out with broadcast bias bo ----------------
__global__ void fill_bias_kernel(float* __restrict__ out, const float* __restrict__ bo) {
  size_t idx = (size_t)blockIdx.x * 256 + threadIdx.x;
  out[idx] = bo[idx & (DM - 1)];
}

// ---------------- Kernel: Q rows GEMM: Q[b,:] = x[b, g(b), :] @ Wq + bq ----------------
__global__ void LB qgemm_kernel(const float* __restrict__ x, const __bf16* __restrict__ WqT,
                                const float* __restrict__ bq, float* __restrict__ Qout) {
  extern __shared__ __align__(16) char smem[];
  __bf16* As = (__bf16*)smem;                  // [64][ASTR]
  const int t = threadIdx.x;
  const int row0 = blockIdx.x * 64;            // batch rows b = row0..row0+63

  // Stage A: gathered rows x[b, b>>7, :] -> bf16 LDS
  #pragma unroll
  for (int i = 0; i < 32; ++i) {
    int idx4 = t + i * 256;                    // 8192 float4 loads
    int r = idx4 >> 7, c4 = idx4 & 127;
    int b = row0 + r;
    int g = b >> 7;
    f32x4 v = *(const f32x4*)(x + (((size_t)b * LVL) + g) * DM + c4 * 4);
    __bf16* dst = &As[r * ASTR + c4 * 4];
    dst[0] = (__bf16)v.x; dst[1] = (__bf16)v.y; dst[2] = (__bf16)v.z; dst[3] = (__bf16)v.w;
  }
  __syncthreads();

  const int w = t >> 5, lane = t & 31;
  const int n0 = w * 64;
  const int nlane = lane & 15, half = lane >> 4;

  v8f acc[4][4];
  #pragma unroll
  for (int mt = 0; mt < 4; ++mt)
    #pragma unroll
    for (int nt = 0; nt < 4; ++nt) acc[mt][nt] = zero8();

  for (int k0 = 0; k0 < DM; k0 += 32) {
    v16bf aF[4];
    #pragma unroll
    for (int mt = 0; mt < 4; ++mt) aF[mt] = load_fragA(&As[(mt * 16) * ASTR + k0], ASTR);
    #pragma unroll
    for (int nt = 0; nt < 4; ++nt) {
      v16bf bF = load_fragB(WqT + (size_t)(n0 + nt * 16) * DM + k0, DM);
      #pragma unroll
      for (int mt = 0; mt < 4; ++mt)
        acc[mt][nt] = wmma_bf16(aF[mt], bF, acc[mt][nt]);
    }
  }

  #pragma unroll
  for (int nt = 0; nt < 4; ++nt) {
    int col = n0 + nt * 16 + nlane;
    float bias = bq[col];
    #pragma unroll
    for (int mt = 0; mt < 4; ++mt) {
      int rbase = mt * 16 + half * 8;
      #pragma unroll
      for (int r = 0; r < 8; ++r)
        Qout[(size_t)(row0 + rbase + r) * DM + col] = acc[mt][nt][r] + bias;
    }
  }
}

// ---------------- Fused attention: K/V WMMA projections + softmax + ctx + p_sum ----------------
__global__ void LB attn_kernel(const float* __restrict__ x,
                               const __bf16* __restrict__ WkT, const __bf16* __restrict__ WvT,
                               const float* __restrict__ bk, const float* __restrict__ bv,
                               const float* __restrict__ Qrow,
                               float* __restrict__ ctx_out, float* __restrict__ psum_out) {
  extern __shared__ __align__(16) char smem[];
  __bf16* As = (__bf16*)smem;                  // [64][ASTR] : x rows for NB batches
  __bf16* Ks = As + 64 * ASTR;                 // [64][ASTR]
  __bf16* Vs = Ks + 64 * ASTR;                 // [64][ASTR]
  float*  qs = (float*)(Vs + 64 * ASTR);       // [NB][512]
  float*  ps = qs + NB * DM;                   // [NB][8][32]

  const int t = threadIdx.x;
  const int b0 = blockIdx.x * NB;

  // Stage A: 64 contiguous x rows starting at (b0, 0)
  const float* xbase = x + (size_t)b0 * LVL * DM;
  #pragma unroll
  for (int i = 0; i < 32; ++i) {
    int idx4 = t + i * 256;
    int r = idx4 >> 7, c4 = idx4 & 127;
    f32x4 v = *(const f32x4*)(xbase + (size_t)r * DM + c4 * 4);
    __bf16* dst = &As[r * ASTR + c4 * 4];
    dst[0] = (__bf16)v.x; dst[1] = (__bf16)v.y; dst[2] = (__bf16)v.z; dst[3] = (__bf16)v.w;
  }
  #pragma unroll
  for (int j = 0; j < 4; ++j) qs[t + j * 256] = Qrow[(size_t)b0 * DM + t + j * 256];
  __syncthreads();

  const int w = t >> 5, lane = t & 31;
  const int n0 = w * 64;
  const int nlane = lane & 15, half = lane >> 4;

  // K then V projection: (2 batches x 32 rows) x 512 = 64x512 output each, K-dim 512
  #pragma unroll 1
  for (int mat = 0; mat < 2; ++mat) {
    const __bf16* WT = mat ? WvT : WkT;
    const float*  bias_v = mat ? bv : bk;
    __bf16* Dst = mat ? Vs : Ks;

    v8f acc[4][4];
    #pragma unroll
    for (int mt = 0; mt < 4; ++mt)
      #pragma unroll
      for (int nt = 0; nt < 4; ++nt) acc[mt][nt] = zero8();

    for (int k0 = 0; k0 < DM; k0 += 32) {
      v16bf aF[4];
      #pragma unroll
      for (int mt = 0; mt < 4; ++mt) aF[mt] = load_fragA(&As[(mt * 16) * ASTR + k0], ASTR);
      #pragma unroll
      for (int nt = 0; nt < 4; ++nt) {
        v16bf bF = load_fragB(WT + (size_t)(n0 + nt * 16) * DM + k0, DM);
        #pragma unroll
        for (int mt = 0; mt < 4; ++mt)
          acc[mt][nt] = wmma_bf16(aF[mt], bF, acc[mt][nt]);
      }
    }

    #pragma unroll
    for (int nt = 0; nt < 4; ++nt) {
      int col = n0 + nt * 16 + nlane;
      float bias = bias_v[col];
      #pragma unroll
      for (int mt = 0; mt < 4; ++mt) {
        int rbase = mt * 16 + half * 8;
        #pragma unroll
        for (int r = 0; r < 8; ++r)
          Dst[(rbase + r) * ASTR + col] = (__bf16)(acc[mt][nt][r] + bias);
      }
    }
  }
  __syncthreads();

  // Scores + softmax: wave w handles head h=w for both batches; lane = key index
  #pragma unroll
  for (int bi = 0; bi < NB; ++bi) {
    const int h = w, kk = lane;
    const float*  qp = qs + bi * DM + h * DH;
    const __bf16* kp = Ks + (bi * LVL + kk) * ASTR + h * DH;
    float s = 0.f;
    #pragma unroll
    for (int d = 0; d < DH; ++d) s += qp[d] * (float)kp[d];
    s *= 0.125f;                                  // 1/sqrt(64)
    float m = s;
    #pragma unroll
    for (int off = 16; off > 0; off >>= 1) m = fmaxf(m, __shfl_xor(m, off, 32));
    float e = __expf(s - m);
    float sum = e;
    #pragma unroll
    for (int off = 16; off > 0; off >>= 1) sum += __shfl_xor(sum, off, 32);
    ps[(bi * NHEAD + h) * 32 + kk] = e / sum;
  }
  __syncthreads();

  // p_attn_sum: mean over heads of the surviving query row
  if (t < NB * LVL) {
    int bi = t >> 5, kk = t & 31;
    float s = 0.f;
    #pragma unroll
    for (int h = 0; h < NHEAD; ++h) s += ps[(bi * NHEAD + h) * 32 + kk];
    psum_out[(size_t)(b0 + bi) * LVL + kk] = s * (1.0f / NHEAD);
  }

  // ctx[n] = sum_k p[h(n)][k] * V[k][n]
  #pragma unroll
  for (int j = 0; j < 4; ++j) {
    int id = t + j * 256;                         // 0..1023
    int bi = id >> 9, n = id & (DM - 1), h = n >> 6;
    const float*  pp = ps + (bi * NHEAD + h) * 32;
    const __bf16* vp = Vs + (bi * LVL) * ASTR + n;
    float a = 0.f;
    #pragma unroll
    for (int kk = 0; kk < LVL; ++kk) a += pp[kk] * (float)vp[kk * ASTR];
    ctx_out[(size_t)(b0 + bi) * DM + n] = a;
  }
}

// ---------------- Output GEMM: out[b, g(b), :] = ctx[b,:] @ Wo + bo ----------------
__global__ void LB ogemm_kernel(const float* __restrict__ ctx, const __bf16* __restrict__ WoT,
                                const float* __restrict__ bo, float* __restrict__ out) {
  extern __shared__ __align__(16) char smem[];
  __bf16* As = (__bf16*)smem;                  // [64][ASTR]
  const int t = threadIdx.x;
  const int row0 = blockIdx.x * 64;

  #pragma unroll
  for (int i = 0; i < 32; ++i) {
    int idx4 = t + i * 256;
    int r = idx4 >> 7, c4 = idx4 & 127;
    f32x4 v = *(const f32x4*)(ctx + (size_t)(row0 + r) * DM + c4 * 4);
    __bf16* dst = &As[r * ASTR + c4 * 4];
    dst[0] = (__bf16)v.x; dst[1] = (__bf16)v.y; dst[2] = (__bf16)v.z; dst[3] = (__bf16)v.w;
  }
  __syncthreads();

  const int w = t >> 5, lane = t & 31;
  const int n0 = w * 64;
  const int nlane = lane & 15, half = lane >> 4;

  v8f acc[4][4];
  #pragma unroll
  for (int mt = 0; mt < 4; ++mt)
    #pragma unroll
    for (int nt = 0; nt < 4; ++nt) acc[mt][nt] = zero8();

  for (int k0 = 0; k0 < DM; k0 += 32) {
    v16bf aF[4];
    #pragma unroll
    for (int mt = 0; mt < 4; ++mt) aF[mt] = load_fragA(&As[(mt * 16) * ASTR + k0], ASTR);
    #pragma unroll
    for (int nt = 0; nt < 4; ++nt) {
      v16bf bF = load_fragB(WoT + (size_t)(n0 + nt * 16) * DM + k0, DM);
      #pragma unroll
      for (int mt = 0; mt < 4; ++mt)
        acc[mt][nt] = wmma_bf16(aF[mt], bF, acc[mt][nt]);
    }
  }

  #pragma unroll
  for (int nt = 0; nt < 4; ++nt) {
    int col = n0 + nt * 16 + nlane;
    float bias = bo[col];
    #pragma unroll
    for (int mt = 0; mt < 4; ++mt) {
      int rbase = mt * 16 + half * 8;
      #pragma unroll
      for (int r = 0; r < 8; ++r) {
        int b = row0 + rbase + r;
        int g = b >> 7;
        out[(((size_t)b * LVL) + g) * DM + col] = acc[mt][nt][r] + bias;
      }
    }
  }
}

extern "C" void kernel_launch(void* const* d_in, const int* in_sizes, int n_in,
                              void* d_out, int out_size, void* d_ws, size_t ws_size,
                              hipStream_t stream) {
  const float* x  = (const float*)d_in[0];
  const float* Wq = (const float*)d_in[1];
  const float* bq = (const float*)d_in[2];
  const float* Wk = (const float*)d_in[3];
  const float* bk = (const float*)d_in[4];
  const float* Wv = (const float*)d_in[5];
  const float* bv = (const float*)d_in[6];
  const float* Wo = (const float*)d_in[7];
  const float* bo = (const float*)d_in[8];

  float* out  = (float*)d_out;                          // [B][L][DM]
  float* psum = out + (size_t)BATCH * LVL * DM;         // [B][L]

  char* ws = (char*)d_ws;
  __bf16* WqT = (__bf16*)ws;
  __bf16* WkT = WqT + (size_t)DM * DM;
  __bf16* WvT = WkT + (size_t)DM * DM;
  __bf16* WoT = WvT + (size_t)DM * DM;
  float*  Q   = (float*)(WoT + (size_t)DM * DM);        // [B][DM]
  float*  ctx = Q + (size_t)BATCH * DM;                 // [B][DM]

  const int GEMM_LDS = 64 * ASTR * 2;                                       // 66,560 B
  const int ATTN_LDS = 3 * 64 * ASTR * 2 + (NB * DM + NB * NHEAD * 32) * 4; // 205,824 B
  (void)hipFuncSetAttribute(reinterpret_cast<const void*>(&qgemm_kernel),
                            hipFuncAttributeMaxDynamicSharedMemorySize, GEMM_LDS);
  (void)hipFuncSetAttribute(reinterpret_cast<const void*>(&ogemm_kernel),
                            hipFuncAttributeMaxDynamicSharedMemorySize, GEMM_LDS);
  (void)hipFuncSetAttribute(reinterpret_cast<const void*>(&attn_kernel),
                            hipFuncAttributeMaxDynamicSharedMemorySize, ATTN_LDS);

  wconvert_kernel<<<(DM * DM) / 256, 256, 0, stream>>>(Wq, Wk, Wv, Wo, WqT, WkT, WvT, WoT);
  qgemm_kernel<<<BATCH / 64, 256, GEMM_LDS, stream>>>(x, WqT, bq, Q);
  fill_bias_kernel<<<(int)(((size_t)BATCH * LVL * DM) / 256), 256, 0, stream>>>(out, bo);
  attn_kernel<<<BATCH / NB, 256, ATTN_LDS, stream>>>(x, WkT, WvT, bk, bv, Q, ctx, psum);
  ogemm_kernel<<<BATCH / 64, 256, GEMM_LDS, stream>>>(ctx, WoT, bo, out);
}
